// MultiHeadSelfAttention_67473936220923
// MI455X (gfx1250) — compile-verified, matrix operand order checked by reading, not run on hardware
//
#include <hip/hip_runtime.h>

typedef unsigned short ushort_t;
typedef __attribute__((ext_vector_type(8)))  unsigned short u16x8;
typedef __attribute__((ext_vector_type(16))) unsigned short u16x16;
typedef __attribute__((ext_vector_type(16))) __bf16         bf16x16;
typedef __attribute__((ext_vector_type(8)))  float          f32x8;
typedef __attribute__((ext_vector_type(4)))  unsigned int   u32x4;
typedef __attribute__((ext_vector_type(8)))  int            i32x8;
typedef __attribute__((ext_vector_type(4)))  int            i32x4;

// ---------------------------------------------------------------- helpers

__device__ __forceinline__ unsigned short f32_to_bf16(float f) {
    union { float f; unsigned u; } v; v.f = f;
    unsigned r = v.u + 0x7FFFu + ((v.u >> 16) & 1u);   // round-to-nearest-even
    return (unsigned short)(r >> 16);
}

__device__ __forceinline__ f32x8 zero8() {
    f32x8 z;
#pragma unroll
    for (int i = 0; i < 8; ++i) z[i] = 0.0f;
    return z;
}

__device__ __forceinline__ bf16x16 make_frag(u16x8 lo, u16x8 hi) {
    u16x16 w;
#pragma unroll
    for (int i = 0; i < 8; ++i) { w[i] = lo[i]; w[i + 8] = hi[i]; }
    return __builtin_bit_cast(bf16x16, w);
}

// Load a 16x32 bf16 A/Bt fragment from a row-major [16][pitch] buffer.
// ISA 16-bit A layout: lane%16 = row, lane/16 selects K-halves
// {h*8 .. h*8+7} and {16+h*8 .. 16+h*8+7} -> two contiguous 16-byte loads.
__device__ __forceinline__ bf16x16 load_frag(const ushort_t* base, int pitch, int lane) {
    const int row = lane & 15;
    const int hi  = (lane >> 4) & 1;
    const ushort_t* p = base + row * pitch + hi * 8;
    u16x8 a = *(const u16x8*)(p);
    u16x8 b = *(const u16x8*)(p + 16);
    return make_frag(a, b);
}

// CDNA5 async global->LDS copy (ASYNCcnt-tracked, bypasses VGPRs).
__device__ __forceinline__ void async_ld_b128(unsigned lds_addr, const void* gaddr) {
    asm volatile("global_load_async_to_lds_b128 %0, %1, off"
                 :: "v"(lds_addr), "v"(gaddr) : "memory");
}
__device__ __forceinline__ void wait_async0() {
    asm volatile("s_wait_asynccnt 0x0" ::: "memory");
}

// CDNA5 Tensor Data Mover: DMA a rows x cols bf16 tile (row stride `stride`
// elements) from global into LDS at lds_addr, padding +4 DWORDs after every
// 16 DWORDs stored -> LDS row pitch 80B (= LDP 40 ushorts).
__device__ __forceinline__ void tdm_load_tile_bf16(unsigned lds_addr, const void* gaddr,
                                                   unsigned rows, unsigned cols,
                                                   unsigned stride,
                                                   unsigned tdim0, unsigned tdim1) {
    unsigned long long ga = (unsigned long long)(uintptr_t)gaddr;
    u32x4 g0;
    g0[0] = 1u;                                      // count=1, user mode
    g0[1] = lds_addr;                                // lds_addr (bytes)
    g0[2] = (unsigned)(ga & 0xFFFFFFFFu);            // global_addr[31:0]
    g0[3] = (unsigned)((ga >> 32) & 0x01FFFFFFu)     // global_addr[56:32]
          | (2u << 30);                              // type = 2 ("image")

    i32x8 g1;
    g1[0] = (int)((1u << 16)        // data_size = 1 -> 2 bytes
                | (1u << 20)        // pad_enable
                | (3u << 22)        // pad_interval: 16 DWORDs
                | (3u << 25));      // pad_amount:   4 DWORDs
    g1[1] = (int)((tdim0 & 0xFFFFu) << 16);                       // tensor_dim0[15:0]
    g1[2] = (int)(((tdim0 >> 16) & 0xFFFFu) | ((tdim1 & 0xFFFFu) << 16));
    g1[3] = (int)(((tdim1 >> 16) & 0xFFFFu) | ((cols & 0xFFFFu) << 16));  // tile_dim0
    g1[4] = (int)(rows & 0xFFFFu);                                // tile_dim1, tile_dim2=0
    g1[5] = (int)stride;                                          // tensor_dim0_stride lo
    g1[6] = 0;
    g1[7] = 0;

    i32x4 gz = {0, 0, 0, 0};
#if __clang_major__ >= 23
    i32x8 gz8 = {0, 0, 0, 0, 0, 0, 0, 0};
    __builtin_amdgcn_tensor_load_to_lds(g0, g1, gz, gz, gz8, 0);
#else
    __builtin_amdgcn_tensor_load_to_lds(g0, g1, gz, gz, 0);
#endif
}

// ---------------------------------------------------------------- converts

__global__ void cvt_f32_bf16(const float* __restrict__ in, ushort_t* __restrict__ out, int n) {
    int i = blockIdx.x * blockDim.x + threadIdx.x;
    if (i < n) out[i] = f32_to_bf16(in[i]);
}

// in: R x Ccols (row-major f32) -> out: Ccols x R (row-major bf16) == in^T
__global__ void cvt_transpose_bf16(const float* __restrict__ in, ushort_t* __restrict__ out,
                                   int R, int Ccols) {
    int i = blockIdx.x * blockDim.x + threadIdx.x;
    if (i >= R * Ccols) return;
    int n = i / R;
    int k = i - n * R;
    out[(size_t)n * R + k] = f32_to_bf16(in[(size_t)k * Ccols + n]);
}

// ---------------------------------------------------------------- GEMM
// D[M,N] = A[M,K] * Bt[N,K]^T + bias[N];  A,Bt bf16, accum f32.
// Block tile 128x128, K-step 32, 8 waves (2x4), 64x32 per wave.
// Double-buffered LDS: A tile staged by the Tensor Data Mover (TENSORcnt),
// B tile staged by async global->LDS copies (ASYNCcnt).

#define TK   32
#define LDP  40   // LDS pitch (ushorts): 80B = 5*16B = TDM pad layout

template <bool BF16_OUT>
__global__ __launch_bounds__(256)
void gemm_bt_wmma(const ushort_t* __restrict__ A, const ushort_t* __restrict__ Bt,
                  const float* __restrict__ bias, void* __restrict__ out,
                  int M, int N, int K) {
    __shared__ ushort_t sA[2][128 * LDP];
    __shared__ ushort_t sB[2][128 * LDP];

    const int tid  = threadIdx.x;
    const int lane = tid & 31;
    const int wave = tid >> 5;
    const int wm   = wave >> 2;   // 0..1
    const int wn   = wave & 3;    // 0..3
    const int bm   = blockIdx.y * 128;
    const int bn   = blockIdx.x * 128;

    auto stage = [&](int buf, int kk) {
        if (wave == 0)   // one TDM descriptor moves the whole 128x32 A tile
            tdm_load_tile_bf16((unsigned)(uintptr_t)&sA[buf][0],
                               &A[(size_t)bm * K + kk],
                               128u, 32u, (unsigned)K,
                               (unsigned)K, (unsigned)(M - bm));
#pragma unroll
        for (int c = 0; c < 2; ++c) {            // B tile: 512 async 16B chunks
            int ch = tid + c * 256;
            int r  = ch >> 2;
            int co = (ch & 3) * 8;
            async_ld_b128((unsigned)(uintptr_t)&sB[buf][r * LDP + co],
                          &Bt[(size_t)(bn + r) * K + kk + co]);
        }
    };
    auto stage_wait = [&]() {
        if (wave == 0) __builtin_amdgcn_s_wait_tensorcnt(0);
        wait_async0();
        __syncthreads();
    };

    f32x8 acc[4][2];
#pragma unroll
    for (int i = 0; i < 4; ++i)
#pragma unroll
        for (int j = 0; j < 2; ++j) acc[i][j] = zero8();

    stage(0, 0);
    stage_wait();

    int cur = 0;
    for (int kk = 0; kk < K; kk += TK) {
        if (kk + TK < K) stage(cur ^ 1, kk + TK);   // overlap next tile with compute

        bf16x16 afr[4];
#pragma unroll
        for (int mi = 0; mi < 4; ++mi)
            afr[mi] = load_frag(&sA[cur][(wm * 64 + mi * 16) * LDP], LDP, lane);
        bf16x16 bfr[2];
#pragma unroll
        for (int ni = 0; ni < 2; ++ni)
            bfr[ni] = load_frag(&sB[cur][(wn * 32 + ni * 16) * LDP], LDP, lane);

#pragma unroll
        for (int mi = 0; mi < 4; ++mi)
#pragma unroll
            for (int ni = 0; ni < 2; ++ni)
                acc[mi][ni] = __builtin_amdgcn_wmma_f32_16x16x32_bf16(
                    false, afr[mi], false, bfr[ni], (short)0, acc[mi][ni], false, false);

        stage_wait();       // DMA writes landed; everyone done reading `cur`
        cur ^= 1;
    }

    // Epilogue: D-layout -> memory. VGPR r: M = r + 8*(lane/16), N = lane%16.
    const int lrow = lane & 15;
    const int lhi  = lane >> 4;
#pragma unroll
    for (int mi = 0; mi < 4; ++mi) {
#pragma unroll
        for (int ni = 0; ni < 2; ++ni) {
            int n = bn + wn * 32 + ni * 16 + lrow;
            float bv = bias[n];
#pragma unroll
            for (int r = 0; r < 8; ++r) {
                int m = bm + wm * 64 + mi * 16 + r + lhi * 8;
                float v = acc[mi][ni][r] + bv;
                if (BF16_OUT) ((ushort_t*)out)[(size_t)m * N + n] = f32_to_bf16(v);
                else          ((float*)   out)[(size_t)m * N + n] = v;
            }
        }
    }
}

// ---------------------------------------------------------------- attention
// Flash-style causal attention. One wave owns a 16-row q tile; iterates keys
// in chunks of 32. S via WMMA (K row-major == Bt), online softmax with
// 16-lane shuffle reductions, P restaged via LDS, V transposed in LDS for PV.

__global__ __launch_bounds__(128)
void attn_wmma(const ushort_t* __restrict__ qkv,   // (B*T) x 3HD bf16 (q|k|v)
               ushort_t* __restrict__ o,           // (B*T) x HD bf16
               int Bn, int T, int Hn, int Dn) {
    __shared__ ushort_t sm[4][16 * 32 + 64 * 32];  // per-wave: P(16x32) + Vt(64x32)
    const int tid  = threadIdx.x;
    const int lane = tid & 31;
    const int wave = tid >> 5;
    ushort_t* pbuf = &sm[wave][0];
    ushort_t* vtb  = &sm[wave][16 * 32];

    const int bh    = blockIdx.y;
    const int b     = bh / Hn;
    const int h     = bh - b * Hn;
    const int qbase = blockIdx.x * 64 + wave * 16;

    const int pitch = 3 * Hn * Dn;                 // 3072
    const ushort_t* qp = qkv + (size_t)(b * T) * pitch + h * Dn;
    const ushort_t* kp = qp + Hn * Dn;
    const ushort_t* vp = qp + 2 * Hn * Dn;

    const int lrow = lane & 15;
    const int lhi  = lane >> 4;

    // Q fragments: 16 rows x 64 d, two K-steps of 32.
    bf16x16 qf[2];
#pragma unroll
    for (int ks = 0; ks < 2; ++ks) {
        const ushort_t* p = qp + (size_t)(qbase + lrow) * pitch + ks * 32 + lhi * 8;
        qf[ks] = make_frag(*(const u16x8*)p, *(const u16x8*)(p + 16));
    }

    float mrow[8], lsum[8];
    f32x8 oacc[4];
#pragma unroll
    for (int r = 0; r < 8; ++r) { mrow[r] = -1e30f; lsum[r] = 0.0f; }
#pragma unroll
    for (int n = 0; n < 4; ++n) oacc[n] = zero8();

    const float sscale = 0.125f;        // 1/sqrt(64)
    const int   kend   = qbase + 16;    // causal extent (exclusive)

    for (int keyb = 0; keyb < kend; keyb += 32) {
        if (keyb + 32 < kend) {         // hint next chunk's K/V rows into cache
            __builtin_prefetch(kp + (size_t)(keyb + 32 + lane) * pitch, 0, 1);
            __builtin_prefetch(vp + (size_t)(keyb + 32 + lane) * pitch, 0, 1);
        }

        // ---- S = Q K^T : two 16-wide key tiles, two d K-steps each.
        f32x8 s[2];
#pragma unroll
        for (int t = 0; t < 2; ++t) {
            s[t] = zero8();
#pragma unroll
            for (int ks = 0; ks < 2; ++ks) {
                const ushort_t* p = kp + (size_t)(keyb + t * 16 + lrow) * pitch + ks * 32 + lhi * 8;
                bf16x16 kf = make_frag(*(const u16x8*)p, *(const u16x8*)(p + 16));
                s[t] = __builtin_amdgcn_wmma_f32_16x16x32_bf16(
                    false, qf[ks], false, kf, (short)0, s[t], false, false);
            }
        }
        // ---- scale + causal mask + chunk row max (row r+8*lhi, col keyb+t*16+lrow)
        float cmax[8];
#pragma unroll
        for (int r = 0; r < 8; ++r) cmax[r] = -1e30f;
#pragma unroll
        for (int t = 0; t < 2; ++t) {
            int col = keyb + t * 16 + lrow;
#pragma unroll
            for (int r = 0; r < 8; ++r) {
                int row = qbase + r + lhi * 8;
                float v = s[t][r] * sscale;
                v = (col <= row) ? v : -1e30f;
                s[t][r] = v;
                cmax[r] = fmaxf(cmax[r], v);
            }
        }
#pragma unroll
        for (int r = 0; r < 8; ++r)
#pragma unroll
            for (int off = 1; off < 16; off <<= 1)
                cmax[r] = fmaxf(cmax[r], __shfl_xor(cmax[r], off, 32));

        // ---- online softmax: P = exp(S - m_new), staged to LDS as bf16
        float mn[8], csum[8];
#pragma unroll
        for (int r = 0; r < 8; ++r) { mn[r] = fmaxf(mrow[r], cmax[r]); csum[r] = 0.0f; }
#pragma unroll
        for (int t = 0; t < 2; ++t) {
            int col = t * 16 + lrow;
#pragma unroll
            for (int r = 0; r < 8; ++r) {
                float p = __expf(s[t][r] - mn[r]);
                csum[r] += p;
                pbuf[(r + lhi * 8) * 32 + col] = f32_to_bf16(p);
            }
        }
#pragma unroll
        for (int r = 0; r < 8; ++r)
#pragma unroll
            for (int off = 1; off < 16; off <<= 1)
                csum[r] += __shfl_xor(csum[r], off, 32);
#pragma unroll
        for (int r = 0; r < 8; ++r) {
            float sc = __expf(mrow[r] - mn[r]);
            lsum[r]  = lsum[r] * sc + csum[r];
            mrow[r]  = mn[r];
#pragma unroll
            for (int n = 0; n < 4; ++n) oacc[n][r] *= sc;
        }

        // ---- stage V^T (64 d x 32 keys) into LDS: lane <-> key
        {
            const ushort_t* p = vp + (size_t)(keyb + lane) * pitch;
#pragma unroll
            for (int dv = 0; dv < 8; ++dv) {
                u16x8 v8 = *(const u16x8*)(p + dv * 8);
#pragma unroll
                for (int e = 0; e < 8; ++e)
                    vtb[(dv * 8 + e) * 32 + lane] = v8[e];
            }
        }

        // ---- O += P V : P as A-frag (K=32 keys), V^T rows as Bt-frags.
        bf16x16 pf = load_frag(pbuf, 32, lane);
#pragma unroll
        for (int n = 0; n < 4; ++n) {
            bf16x16 vf = load_frag(&vtb[(n * 16) * 32], 32, lane);
            oacc[n] = __builtin_amdgcn_wmma_f32_16x16x32_bf16(
                false, pf, false, vf, (short)0, oacc[n], false, false);
        }
    }

    // ---- finalize: O /= l, store bf16 at [b*T+row, h*D + d]
#pragma unroll
    for (int n = 0; n < 4; ++n)
#pragma unroll
        for (int r = 0; r < 8; ++r) {
            int row = qbase + r + lhi * 8;
            int col = h * Dn + n * 16 + lrow;
            float v = oacc[n][r] / lsum[r];
            o[(size_t)(b * T + row) * (Hn * Dn) + col] = f32_to_bf16(v);
        }
}

// ---------------------------------------------------------------- launcher

extern "C" void kernel_launch(void* const* d_in, const int* in_sizes, int n_in,
                              void* d_out, int out_size, void* d_ws, size_t ws_size,
                              hipStream_t stream) {
    const float* x     = (const float*)d_in[0];
    const float* w_qkv = (const float*)d_in[1];
    const float* b_qkv = (const float*)d_in[2];
    const float* w_out = (const float*)d_in[3];
    const float* b_out = (const float*)d_in[4];
    float* out = (float*)d_out;

    const int Bn = 2, T = 2048, C = 1024, H = 16, D = 64;
    const int MT = Bn * T;      // 4096 rows
    const int N3 = 3 * C;       // 3072

    char* ws = (char*)d_ws;
    ushort_t* xb    = (ushort_t*)ws;  ws += (size_t)MT * C  * 2;   //  8 MB
    ushort_t* wqT   = (ushort_t*)ws;  ws += (size_t)N3 * C  * 2;   //  6 MB
    ushort_t* qkvb  = (ushort_t*)ws;  ws += (size_t)MT * N3 * 2;   // 24 MB
    ushort_t* attnb = (ushort_t*)ws;  ws += (size_t)MT * C  * 2;   //  8 MB
    ushort_t* woT   = (ushort_t*)ws;                               //  2 MB

    cvt_f32_bf16<<<(MT * C + 255) / 256, 256, 0, stream>>>(x, xb, MT * C);
    cvt_transpose_bf16<<<(C * N3 + 255) / 256, 256, 0, stream>>>(w_qkv, wqT, C, N3);
    cvt_transpose_bf16<<<(C * C + 255) / 256, 256, 0, stream>>>(w_out, woT, C, C);

    // qkv = x @ w_qkv + b_qkv   (bf16 out)
    gemm_bt_wmma<true><<<dim3(N3 / 128, MT / 128), 256, 0, stream>>>(
        xb, wqT, b_qkv, (void*)qkvb, MT, N3, C);

    // causal multi-head attention
    attn_wmma<<<dim3(T / 64, Bn * H), 128, 0, stream>>>(qkvb, attnb, Bn, T, H, D);

    // out = attn @ w_out + b_out  (f32 out)
    gemm_bt_wmma<false><<<dim3(C / 128, MT / 128), 256, 0, stream>>>(
        attnb, woT, b_out, (void*)out, MT, C, C);
}
